// RWKVLayer_62947040690528
// MI455X (gfx1250) — matrix operand order repaired and checked by reading, not compiled, not run on hardware
//
#include <hip/hip_runtime.h>
#include <hip/hip_bf16.h>
#include <math.h>
#include <stdint.h>

typedef __bf16 bf16;
typedef __attribute__((ext_vector_type(16))) __bf16 v16bf;
typedef __attribute__((ext_vector_type(8)))  __bf16 v8bf;
typedef __attribute__((ext_vector_type(8)))  float  v8f;

// GCC-style int4 vector + address-space-qualified pointee types matching the
// builtin prototype vV4i*1V4i*3IiIi (global src, LDS dst, imm offset, imm cpol)
typedef int v4i __attribute__((vector_size(16)));
typedef __attribute__((address_space(1))) v4i global_v4i;
typedef __attribute__((address_space(3))) v4i shared_v4i;

// CDNA5 async global->LDS path (ASYNCcnt), guarded so the file compiles on
// toolchains where the builtin is absent (fallback: plain LDS staging).
#if defined(__HIP_DEVICE_COMPILE__) && defined(__has_builtin)
#  if __has_builtin(__builtin_amdgcn_global_load_async_to_lds_b128)
#    define HAVE_ASYNC_LDS 1
#  else
#    define HAVE_ASYNC_LDS 0
#  endif
#else
#  define HAVE_ASYNC_LDS 0
#endif

#define LDS_PAD 40  // row stride in bf16 elems: 80B, 16B-aligned, bank-spread

__device__ __forceinline__ void async_wait0() {
#if HAVE_ASYNC_LDS
#  if __has_builtin(__builtin_amdgcn_s_wait_asynccnt)
  __builtin_amdgcn_s_wait_asynccnt(0);
#  else
  asm volatile("s_wait_asynccnt 0x0" ::: "memory");
#  endif
#endif
}

__device__ __forceinline__ void async_copy16(const bf16* g, bf16* l) {
#if HAVE_ASYNC_LDS
  __builtin_amdgcn_global_load_async_to_lds_b128(
      (global_v4i*)(uintptr_t)g,
      (shared_v4i*)(unsigned int)(uintptr_t)l, 0, 0);
#else
  *(v8bf*)l = *(const v8bf*)g;
#endif
}

// ---------------------------------------------------------------------------
// f32 -> bf16 conversion (weights)
// ---------------------------------------------------------------------------
__global__ __launch_bounds__(256) void cvt_f32_bf16(const float* __restrict__ src,
                                                    bf16* __restrict__ dst, int n) {
  int i = blockIdx.x * blockDim.x + threadIdx.x;
  if (i < n) dst[i] = (bf16)src[i];
}

// ---------------------------------------------------------------------------
// time-shift mixing: xk/xv/xr in bf16
// ---------------------------------------------------------------------------
__global__ __launch_bounds__(256) void timemix_kernel(
    const float* __restrict__ x,
    const float* __restrict__ mk, const float* __restrict__ mv,
    const float* __restrict__ mr,
    bf16* __restrict__ xk, bf16* __restrict__ xv, bf16* __restrict__ xr,
    int total, int D, int TD) {
  int i = blockIdx.x * blockDim.x + threadIdx.x;
  if (i >= total) return;
  float xc = x[i];
  float lx = ((i % TD) >= D) ? x[i - D] : 0.f;  // t==0 -> zero init buffer
  int d = i % D;
  float k = mk[d], v = mv[d], r = mr[d];
  xk[i] = (bf16)(xc * k + lx * (1.f - k));
  xv[i] = (bf16)(xc * v + lx * (1.f - v));
  xr[i] = (bf16)(xc * r + lx * (1.f - r));
}

// ---------------------------------------------------------------------------
// Stage one 128x32 bf16 tile of A and of W into LDS.
// gA/gW point at (block_row0, kb). 256 threads x 2 chunks x 16B each per tile.
// ---------------------------------------------------------------------------
__device__ __forceinline__ void stage_tiles(const bf16* __restrict__ gA,
                                            const bf16* __restrict__ gW,
                                            bf16* __restrict__ la,
                                            bf16* __restrict__ lw,
                                            int tid, int K) {
#pragma unroll
  for (int i = 0; i < 2; ++i) {
    int c   = tid + 256 * i;   // 0..511 chunk id
    int row = c >> 2;          // 0..127
    int col = (c & 3) * 8;     // 0,8,16,24 (bf16 elems) = 16B chunks
    async_copy16(gA + (size_t)row * K + col, la + row * LDS_PAD + col);
    async_copy16(gW + (size_t)row * K + col, lw + row * LDS_PAD + col);
  }
}

// ---------------------------------------------------------------------------
// bf16 WMMA GEMM:  C[m,n] = sum_k A[m,k] * W[n,k]   (einsum 'btd,ed->bte')
// A: (M,K) bf16 row-major,  W: (N,K) bf16 row-major.
// Block = 8 waves, block tile 128x128, wave tile 32x64 (2x4 16x16 accum).
// K-step 32, double-buffered LDS staging via CDNA5 async global->LDS copies.
// MODE 0: store f32.  MODE 1: store bf16(sigmoid(x)).
// ---------------------------------------------------------------------------
template <int MODE>
__global__ __launch_bounds__(256) void gemm_wmma_bf16(
    const bf16* __restrict__ A, const bf16* __restrict__ W,
    void* __restrict__ Cout, int M, int N, int K) {
  __shared__ bf16 sA[2][128 * LDS_PAD];
  __shared__ bf16 sW[2][128 * LDS_PAD];

  const int tid  = threadIdx.x;
  const int lane = tid & 31;
  const int wave = tid >> 5;
  const int half = lane >> 4;   // lane group 0-15 vs 16-31
  const int l16  = lane & 15;

  const int wn = (wave & 1) * 64;        // wave N offset in block tile
  const int wm = (wave >> 1) * 32;       // wave M offset in block tile
  const int block_n = blockIdx.x * 128;
  const int block_m = blockIdx.y * 128;

  const bf16* Ablk = A + (size_t)block_m * K;
  const bf16* Wblk = W + (size_t)block_n * K;

  v8f acc[2][4];
#pragma unroll
  for (int i = 0; i < 2; ++i)
#pragma unroll
    for (int j = 0; j < 4; ++j) {
      v8f z = {0.f, 0.f, 0.f, 0.f, 0.f, 0.f, 0.f, 0.f};
      acc[i][j] = z;
    }

  // prologue: stage K-tile 0 into buffer 0
  stage_tiles(Ablk, Wblk, sA[0], sW[0], tid, K);
  async_wait0();
  __syncthreads();

  int buf = 0;
  for (int kb = 0; kb < K; kb += 32) {
    if (kb + 32 < K)  // overlap: async-copy next K tile while computing
      stage_tiles(Ablk + kb + 32, Wblk + kb + 32, sA[buf ^ 1], sW[buf ^ 1],
                  tid, K);
    if (kb + 64 < K) {  // warm L2 two tiles ahead (global_prefetch_b8)
      __builtin_prefetch(Ablk + (size_t)(l16 * 8) * K + kb + 64, 0, 2);
      __builtin_prefetch(Wblk + (size_t)(l16 * 8) * K + kb + 64, 0, 2);
    }

    // A fragments (16x32): lanes 0-15 hold K {0..7,16..23}, lanes 16-31 +8
    const bf16* sa = sA[buf];
    const bf16* sw = sW[buf];
    v16bf afrag[2];
#pragma unroll
    for (int i = 0; i < 2; ++i) {
      const bf16* ap = sa + (wm + i * 16 + l16) * LDS_PAD + half * 8;
      v8bf lo = *(const v8bf*)(ap);
      v8bf hi = *(const v8bf*)(ap + 16);
      v16bf a;
#pragma unroll
      for (int e = 0; e < 8; ++e) { a[e] = lo[e]; a[e + 8] = hi[e]; }
      afrag[i] = a;
    }
    // B fragments (32x16): lane n=l16; lanes 0-15 K {0..7,16..23}, lanes 16-31 +8
    v16bf bfrag[4];
#pragma unroll
    for (int j = 0; j < 4; ++j) {
      const bf16* bp = sw + (wn + j * 16 + l16) * LDS_PAD + half * 8;
      v8bf lo = *(const v8bf*)(bp);
      v8bf hi = *(const v8bf*)(bp + 16);
      v16bf b;
#pragma unroll
      for (int e = 0; e < 8; ++e) { b[e] = lo[e]; b[e + 8] = hi[e]; }
      bfrag[j] = b;
    }
#pragma unroll
    for (int i = 0; i < 2; ++i)
#pragma unroll
      for (int j = 0; j < 4; ++j)
        acc[i][j] = __builtin_amdgcn_wmma_f32_16x16x32_bf16(
            false, afrag[i], false, bfrag[j], (short)0, acc[i][j], false, false);

    async_wait0();     // our async copies into buf^1 done
    __syncthreads();   // everyone's copies + LDS reads done
    buf ^= 1;
  }

  // Epilogue: C/D layout -> row = vgpr + 8*half, col = l16
#pragma unroll
  for (int i = 0; i < 2; ++i)
#pragma unroll
    for (int j = 0; j < 4; ++j)
#pragma unroll
      for (int r = 0; r < 8; ++r) {
        int m = block_m + wm + i * 16 + half * 8 + r;
        int n = block_n + wn + j * 16 + l16;
        float val = acc[i][j][r];
        if (MODE == 0) {
          ((float*)Cout)[(size_t)m * N + n] = val;
        } else {
          ((bf16*)Cout)[(size_t)m * N + n] = (bf16)(1.f / (1.f + __expf(-val)));
        }
      }
}

// ---------------------------------------------------------------------------
// log-space WKV scan: serial over T, one thread per (b,d) channel.
// 8192 channels = 256 wave32's across the device.
// ---------------------------------------------------------------------------
__device__ __forceinline__ float lae(float a, float b) {  // logaddexp
  float m = fmaxf(a, b);
  if (m == -INFINITY) return -INFINITY;
  return m + log1pf(__expf(fminf(a, b) - m));
}

__global__ __launch_bounds__(256) void wkv_scan_kernel(
    const float* __restrict__ kbuf, const float* __restrict__ vbuf,
    const bf16* __restrict__ sr, const float* __restrict__ time_decay,
    const float* __restrict__ time_first, bf16* __restrict__ rwkv,
    int B, int T, int D) {
  int gid = blockIdx.x * blockDim.x + threadIdx.x;
  if (gid >= B * D) return;
  int b = gid / D, d = gid % D;
  float w = __expf(time_decay[d]);
  float u = time_first[d];
  float ln_ap = -INFINITY, ln_am = -INFINITY, ln_b = -INFINITY;
  size_t base = (size_t)b * T * D + d;
  for (int t = 0; t < T; ++t) {
    size_t off = base + (size_t)t * D;
    float kt = kbuf[off];
    float vt = vbuf[off];
    float ln_vp = __logf(fmaxf(vt, 0.f) + 1e-4f);
    float ln_vm = __logf(fmaxf(-vt, 0.f) + 1e-4f);
    float uk = u + kt;
    float denom = lae(uk, ln_b);
    float wkv = __expf(lae(uk + ln_vp, ln_ap) - denom) -
                __expf(lae(uk + ln_vm, ln_am) - denom);
    rwkv[off] = (bf16)(wkv * (float)sr[off]);
    ln_ap = lae(ln_ap - w, kt + ln_vp);
    ln_am = lae(ln_am - w, kt + ln_vm);
    ln_b  = lae(ln_b  - w, kt);
  }
}

// ---------------------------------------------------------------------------
extern "C" void kernel_launch(void* const* d_in, const int* in_sizes, int n_in,
                              void* d_out, int out_size, void* d_ws, size_t ws_size,
                              hipStream_t stream) {
  const float* x  = (const float*)d_in[0];
  const float* td = (const float*)d_in[1];
  const float* tf = (const float*)d_in[2];
  const float* mk = (const float*)d_in[3];
  const float* mv = (const float*)d_in[4];
  const float* mr = (const float*)d_in[5];
  const float* Wk = (const float*)d_in[6];
  const float* Wv = (const float*)d_in[7];
  const float* Wr = (const float*)d_in[8];
  const float* Wo = (const float*)d_in[9];

  const int D   = in_sizes[1];          // 1024
  const int BTD = in_sizes[0];          // B*T*D
  const int B   = 8;
  const int T   = BTD / (B * D);        // 2048
  const int M   = B * T;                // 16384

  // workspace layout
  char* ws = (char*)d_ws;
  const size_t bt16 = (size_t)BTD * 2;  // bf16 activation buffer
  const size_t bt32 = (size_t)BTD * 4;  // f32 activation buffer
  const size_t w16  = (size_t)D * D * 2;
  size_t off = 0;
  bf16*  xk   = (bf16*)(ws + off); off += bt16;
  bf16*  xv   = (bf16*)(ws + off); off += bt16;
  bf16*  xr   = (bf16*)(ws + off); off += bt16;
  bf16*  Wk16 = (bf16*)(ws + off); off += w16;
  bf16*  Wv16 = (bf16*)(ws + off); off += w16;
  bf16*  Wr16 = (bf16*)(ws + off); off += w16;
  bf16*  Wo16 = (bf16*)(ws + off); off += w16;
  float* kbuf = (float*)(ws + off); off += bt32;
  float* vbuf = (float*)(ws + off); off += bt32;
  bf16*  sr   = (bf16*)(ws + off); off += bt16;
  bf16*  rwkv = xk;  // xk dead after k-GEMM; reuse

  // 1) weights -> bf16
  {
    int n = D * D;
    int g = (n + 255) / 256;
    cvt_f32_bf16<<<g, 256, 0, stream>>>(Wk, Wk16, n);
    cvt_f32_bf16<<<g, 256, 0, stream>>>(Wv, Wv16, n);
    cvt_f32_bf16<<<g, 256, 0, stream>>>(Wr, Wr16, n);
    cvt_f32_bf16<<<g, 256, 0, stream>>>(Wo, Wo16, n);
  }

  // 2) time-shift mixing
  {
    int g = (BTD + 255) / 256;
    timemix_kernel<<<g, 256, 0, stream>>>(x, mk, mv, mr, xk, xv, xr,
                                          BTD, D, T * D);
  }

  // 3) projections: k, v (f32), sr = sigmoid(r) (bf16)
  {
    dim3 grid(D / 128, M / 128);
    gemm_wmma_bf16<0><<<grid, 256, 0, stream>>>(xk, Wk16, kbuf, M, D, D);
    gemm_wmma_bf16<0><<<grid, 256, 0, stream>>>(xv, Wv16, vbuf, M, D, D);
    gemm_wmma_bf16<1><<<grid, 256, 0, stream>>>(xr, Wr16, sr,   M, D, D);
  }

  // 4) serial-T WKV scan, fused * sigmoid(r), bf16 out
  {
    int g = (B * D + 255) / 256;
    wkv_scan_kernel<<<g, 256, 0, stream>>>(kbuf, vbuf, sr, td, tf, rwkv, B, T, D);
  }

  // 5) output projection -> f32 d_out
  {
    dim3 grid(D / 128, M / 128);
    gemm_wmma_bf16<0><<<grid, 256, 0, stream>>>(rwkv, Wo16, d_out, M, D, D);
  }
}
